// ResidualGRUBlock_31061203485107
// MI455X (gfx1250) — matrix operand order repaired and checked by reading, not compiled, not run on hardware
//
#include <hip/hip_runtime.h>
#include <hip/hip_bf16.h>

typedef __bf16 bf16_t;
typedef __attribute__((ext_vector_type(16))) __bf16 v16bf;
typedef __attribute__((ext_vector_type(8)))  __bf16 v8bf;
typedef __attribute__((ext_vector_type(8)))  float  v8f;
typedef __attribute__((ext_vector_type(4)))  unsigned int u32x4;
typedef __attribute__((ext_vector_type(8)))  int i32x8;
typedef __attribute__((ext_vector_type(4)))  int i32x4;

#define TT    1024   // sequence length
#define BBB   64     // batch
#define FF    256    // input features
#define UUU   512    // GRU units
#define NU    1536   // 3*U
#define NPROJ 4096   // 1536 (fwd gates) + 1536 (bwd gates) + 1024 (residual)

union CFrag { v8f v; float f[8]; };
union AFrag { v16bf v; v8bf h[2]; };

// ---- WMMA fragment loaders (wave32; A and B both use contiguous k = 16*(lane/16) + s) ----------
__device__ __forceinline__ v16bf wmma_ld_a(const bf16_t* base, int stride, int kt, int lane) {
    const int r  = lane & 15;
    const int kb = kt * 32 + ((lane >> 4) << 4);
    const bf16_t* p = base + (size_t)r * stride + kb;
    AFrag a;
    a.h[0] = *(const v8bf*)(p);
    a.h[1] = *(const v8bf*)(p + 8);
    return a.v;
}

__device__ __forceinline__ v16bf wmma_ld_b(const bf16_t* pack, int frag, int lane) {
    return *(const v16bf*)(pack + (((size_t)frag * 32 + lane) << 4));
}

__device__ __forceinline__ v8f wmma_bf16(v16bf a, v16bf b, v8f c) {
    return __builtin_amdgcn_wmma_f32_16x16x32_bf16(false, a, false, b, (short)0, c, false, false);
}

__device__ __forceinline__ float sigm(float x) { return 1.f / (1.f + __expf(-x)); }

// ---- TDM: one 2D tile (tile_d1 rows x tile_d0 f32) global -> LDS -----------
// Descriptor per CDNA5 ISA (2D: groups 2/3 zero).
__device__ __forceinline__ void tdm_load_tile_f32(unsigned lds_off,
                                                  unsigned long long gaddr,
                                                  unsigned tensor_d0, unsigned tensor_d1,
                                                  unsigned tile_d0, unsigned tile_d1,
                                                  unsigned long long row_stride_elems) {
    u32x4 g0;
    g0[0] = 1u;                                            // count=1, user descriptor
    g0[1] = lds_off;                                       // lds_addr (bytes)
    g0[2] = (unsigned)(gaddr & 0xFFFFFFFFu);               // global_addr[31:0]
    g0[3] = (unsigned)((gaddr >> 32) & 0x01FFFFFFu)        // global_addr[56:32]
          | (2u << 30);                                    // type = 2 ("image")
    i32x8 g1;
    g1[0] = (int)(2u << 16);                               // data_size=2 -> 4 bytes
    g1[1] = (int)((tensor_d0 & 0xFFFFu) << 16);            // tensor_dim0[15:0]
    g1[2] = (int)(((tensor_d0 >> 16) & 0xFFFFu)
          | ((tensor_d1 & 0xFFFFu) << 16));                // dim0[31:16] | dim1[15:0]
    g1[3] = (int)(((tensor_d1 >> 16) & 0xFFFFu)
          | ((tile_d0 & 0xFFFFu) << 16));                  // dim1[31:16] | tile_dim0
    g1[4] = (int)(tile_d1 & 0xFFFFu);                      // tile_dim1 | tile_dim2=0
    g1[5] = (int)(row_stride_elems & 0xFFFFFFFFu);         // tensor_dim0_stride[31:0]
    g1[6] = (int)((row_stride_elems >> 32) & 0xFFFFu);     // stride[47:32] | dim1_stride lo
    g1[7] = 0;
    i32x4 z4 = {0, 0, 0, 0};
#if defined(__clang_major__) && (__clang_major__ >= 23)
    i32x8 z8 = {0, 0, 0, 0, 0, 0, 0, 0};
    __builtin_amdgcn_tensor_load_to_lds(g0, g1, z4, z4, z8, 0);
#else
    __builtin_amdgcn_tensor_load_to_lds(g0, g1, z4, z4, 0);
#endif
}

// ---- prep: fp32 -> bf16 copy of x ------------------------------------------
__global__ __launch_bounds__(256) void cvt_x_kernel(const float* __restrict__ x,
                                                    bf16_t* __restrict__ xbf) {
    size_t i = (size_t)blockIdx.x * 256 + threadIdx.x;
    xbf[i] = (bf16_t)x[i];
}

// ---- prep: pack [Wf | Wb | Wp] (K=256, N=4096) into B-fragment layout ------
__global__ __launch_bounds__(256) void pack_w_kernel(const float* __restrict__ Wf,
                                                     const float* __restrict__ Wb,
                                                     const float* __restrict__ Wp,
                                                     bf16_t* __restrict__ out) {
    size_t idx = (size_t)blockIdx.x * 256 + threadIdx.x;   // NT=256, KT=8
    int i    = idx & 15;
    int lane = (idx >> 4) & 31;
    int kt   = (idx >> 9) & 7;
    int nt   = (int)(idx >> 12);
    int k = kt * 32 + ((lane >> 4) << 4) + i;
    int n = nt * 16 + (lane & 15);
    float v;
    if (n < 1536)      v = Wf[(size_t)k * 1536 + n];
    else if (n < 3072) v = Wb[(size_t)k * 1536 + (n - 1536)];
    else               v = Wp[(size_t)k * 1024 + (n - 3072)];
    out[idx] = (bf16_t)v;
}

// ---- prep: pack U (K=512, N=1536) into B-fragment layout -------------------
__global__ __launch_bounds__(256) void pack_u_kernel(const float* __restrict__ U,
                                                     bf16_t* __restrict__ out) {
    size_t idx = (size_t)blockIdx.x * 256 + threadIdx.x;   // NT=96, KT=16
    int i    = idx & 15;
    int lane = (idx >> 4) & 31;
    int kt   = (idx >> 9) & 15;
    int nt   = (int)(idx >> 13);
    int k = kt * 32 + ((lane >> 4) << 4) + i;
    int n = nt * 16 + (lane & 15);
    out[idx] = (bf16_t)U[(size_t)k * 1536 + n];
}

// ---- big time-parallel GEMM: proj = x @ [Wf|Wb|Wp] + [bi_f|bi_b|0] ---------
__global__ __launch_bounds__(256) void proj_kernel(const bf16_t* __restrict__ xbf,
                                                   const bf16_t* __restrict__ wpack,
                                                   const float* __restrict__ bfv,
                                                   const float* __restrict__ bbv,
                                                   float* __restrict__ proj) {
    const int mt   = blockIdx.x;
    const int wave = threadIdx.x >> 5;
    const int lane = threadIdx.x & 31;
    const int nt0  = blockIdx.y * 32 + wave * 4;

    CFrag c[4] = {};
    const bf16_t* abase = xbf + (size_t)mt * 16 * FF;

    for (int kt = 0; kt < 8; ++kt) {
        v16bf a = wmma_ld_a(abase, FF, kt, lane);
        if (kt + 1 < 8)   // pull next A-chunk toward L0 (global_prefetch_b8)
            __builtin_prefetch(abase + (size_t)(lane & 15) * FF + (kt + 1) * 32, 0, 1);
#pragma unroll
        for (int q = 0; q < 4; ++q) {
            v16bf b = wmma_ld_b(wpack, (nt0 + q) * 8 + kt, lane);
            c[q].v = wmma_bf16(a, b, c[q].v);
        }
    }

    const int rbase = (lane >> 4) << 3;
#pragma unroll
    for (int q = 0; q < 4; ++q) {
        const int col = (nt0 + q) * 16 + (lane & 15);
        const float bias = (col < 1536) ? bfv[col]
                         : (col < 3072) ? bbv[col - 1536] : 0.f;
#pragma unroll
        for (int j = 0; j < 8; ++j) {
            const int row = mt * 16 + rbase + j;
            proj[(size_t)row * NPROJ + col] = c[q].f[j] + bias;
        }
    }
}

// ---- sequential recurrence: 8 persistent WGs (4 batch-tiles x 2 dirs) ------
// LDS (dynamic, 144 KB of the WGP's 320 KB):
//   [0,32K)   h_f   : fp32 state, 16 x 512
//   [32K,48K) h_bf  : bf16 state (WMMA A operand), 16 x 512
//   [48K,144K) stage: per-step gate preactivations, 16 x 1536 f32, TDM-filled
__global__ __launch_bounds__(512) void gru_kernel(const float* __restrict__ proj,
                                                  const bf16_t* __restrict__ upackf,
                                                  const bf16_t* __restrict__ upackb,
                                                  const float* __restrict__ bfv,
                                                  const float* __restrict__ bbv,
                                                  float* __restrict__ gruY) {
    extern __shared__ char smem[];
    float*  h_f   = (float*)smem;                  // 32 KB
    bf16_t* h_bf  = (bf16_t*)(smem + 32768);       // 16 KB
    float*  stage = (float*)(smem + 49152);        // 96 KB

    const int dir = blockIdx.x >> 2;
    const int b0  = (blockIdx.x & 3) * 16;
    const int tid = threadIdx.x, wave = tid >> 5, lane = tid & 31;

    for (int i = tid; i < 16 * UUU; i += 512) { h_f[i] = 0.f; h_bf[i] = (bf16_t)0.f; }
    __syncthreads();
    __builtin_amdgcn_s_cluster_barrier();   // NOP when not dispatched as a cluster

    const bf16_t* up  = dir ? upackb : upackf;
    const float*  br  = (dir ? bbv : bfv) + 1536;   // b[1] = recurrent bias
    const int u0      = wave * 32;
    const int dbase   = dir ? 1536 : 0;
    const int rbase   = (lane >> 4) << 3;
    const unsigned stage_off = (unsigned)(unsigned long long)(size_t)stage; // LDS byte offset

    float brz[2], brr[2], brh[2];
#pragma unroll
    for (int p = 0; p < 2; ++p) {
        int u = u0 + 16 * p + (lane & 15);
        brz[p] = br[u]; brr[p] = br[512 + u]; brh[p] = br[1024 + u];
    }

    for (int s = 0; s < TT; ++s) {
        const int t = dir ? (TT - 1 - s) : s;

        // DMA this step's 16 x 1536 f32 gate tile into LDS; overlaps the GEMM.
        if (tid < 32) {
            const unsigned long long gaddr = (unsigned long long)(size_t)proj
                + (((unsigned long long)b0 * TT + (unsigned)t) * NPROJ + (unsigned)dbase) * 4ull;
            tdm_load_tile_f32(stage_off, gaddr,
                              /*tensor_d0=*/NU, /*tensor_d1=*/16,
                              /*tile_d0=*/NU,   /*tile_d1=*/16,
                              /*row_stride=*/(unsigned long long)TT * NPROJ);
        }

        CFrag cz[2] = {}, cr[2] = {}, ch[2] = {};
        for (int kt = 0; kt < 16; ++kt) {
            v16bf a = wmma_ld_a(h_bf, UUU, kt, lane);    // ds_load_b128 x2
#pragma unroll
            for (int p = 0; p < 2; ++p) {
                const int ntz = 2 * wave + p;
                cz[p].v = wmma_bf16(a, wmma_ld_b(up, ntz * 16 + kt, lane),        cz[p].v);
                cr[p].v = wmma_bf16(a, wmma_ld_b(up, (32 + ntz) * 16 + kt, lane), cr[p].v);
                ch[p].v = wmma_bf16(a, wmma_ld_b(up, (64 + ntz) * 16 + kt, lane), ch[p].v);
            }
        }

        if (tid < 32) __builtin_amdgcn_s_wait_tensorcnt(0);
        __syncthreads();   // stage ready; all cross-wave h_bf reads complete

        // Each wave touches only its own 32-col slice of h_f/h_bf: no intra-step hazard.
#pragma unroll
        for (int p = 0; p < 2; ++p) {
            const int u = u0 + 16 * p + (lane & 15);
#pragma unroll
            for (int j = 0; j < 8; ++j) {
                const int rm = rbase + j;
                const float xz = stage[rm * NU + u];
                const float xr = stage[rm * NU + 512 + u];
                const float xh = stage[rm * NU + 1024 + u];
                const float z  = sigm(xz + cz[p].f[j] + brz[p]);
                const float r  = sigm(xr + cr[p].f[j] + brr[p]);
                const float hv = tanhf(xh + r * (ch[p].f[j] + brh[p]));
                const float hn = z * h_f[rm * UUU + u] + (1.f - z) * hv;
                h_f [rm * UUU + u] = hn;
                h_bf[rm * UUU + u] = (bf16_t)hn;
                gruY[((size_t)(b0 + rm) * TT + t) * 1024 + dir * 512 + u] = hn;
            }
        }
        __syncthreads();   // h_bf + stage safe for next step's GEMM / TDM
    }
}

// ---- LayerNorm epilogue: y = gru_out + residual; normalize over 1024 cols --
__global__ __launch_bounds__(256) void ln_kernel(const float* __restrict__ gruY,
                                                 const float* __restrict__ proj,
                                                 const float* __restrict__ gamma,
                                                 const float* __restrict__ beta,
                                                 float* __restrict__ out) {
    const int row = blockIdx.x;
    const int tid = threadIdx.x;
    float y[4], s = 0.f, s2 = 0.f;
#pragma unroll
    for (int q = 0; q < 4; ++q) {
        const int c = tid + q * 256;
        const float v = gruY[(size_t)row * 1024 + c] + proj[(size_t)row * NPROJ + 3072 + c];
        y[q] = v; s += v; s2 += v * v;
    }
#pragma unroll
    for (int off = 16; off; off >>= 1) {
        s  += __shfl_xor(s,  off, 32);
        s2 += __shfl_xor(s2, off, 32);
    }
    __shared__ float ws1[8], ws2[8];
    const int wave = tid >> 5, lane = tid & 31;
    if (lane == 0) { ws1[wave] = s; ws2[wave] = s2; }
    __syncthreads();
    if (tid == 0) {
        float a = 0.f, b = 0.f;
        for (int i = 0; i < 8; ++i) { a += ws1[i]; b += ws2[i]; }
        ws1[0] = a; ws2[0] = b;
    }
    __syncthreads();
    const float mu  = ws1[0] * (1.f / 1024.f);
    const float var = ws2[0] * (1.f / 1024.f) - mu * mu;
    const float inv = rsqrtf(var + 1e-3f);
#pragma unroll
    for (int q = 0; q < 4; ++q) {
        const int c = tid + q * 256;
        out[(size_t)row * 1024 + c] = gamma[c] * (y[q] - mu) * inv + beta[c];
    }
}

extern "C" void kernel_launch(void* const* d_in, const int* in_sizes, int n_in,
                              void* d_out, int out_size, void* d_ws, size_t ws_size,
                              hipStream_t stream) {
    (void)in_sizes; (void)n_in; (void)out_size; (void)ws_size;
    const float* x     = (const float*)d_in[0];
    const float* Wf    = (const float*)d_in[1];
    const float* Uf    = (const float*)d_in[2];
    const float* bfv   = (const float*)d_in[3];
    const float* Wb    = (const float*)d_in[4];
    const float* Ub    = (const float*)d_in[5];
    const float* bbv   = (const float*)d_in[6];
    const float* Wp    = (const float*)d_in[7];
    const float* gamma = (const float*)d_in[8];
    const float* beta  = (const float*)d_in[9];
    float* out = (float*)d_out;

    char* ws = (char*)d_ws;
    size_t o = 0;
    bf16_t* xbf   = (bf16_t*)(ws + o); o += (size_t)BBB * TT * FF * 2;     // 32 MB
    bf16_t* wpack = (bf16_t*)(ws + o); o += (size_t)FF * NPROJ * 2;        // 2 MB
    bf16_t* upf   = (bf16_t*)(ws + o); o += (size_t)UUU * NU * 2;          // 1.5 MB
    bf16_t* upb   = (bf16_t*)(ws + o); o += (size_t)UUU * NU * 2;          // 1.5 MB
    float*  proj  = (float*) (ws + o); o += (size_t)BBB * TT * NPROJ * 4;  // 1 GB
    float*  gruY  = (float*) (ws + o);                                     // 256 MB

    cvt_x_kernel <<<(BBB * TT * FF) / 256, 256, 0, stream>>>(x, xbf);
    pack_w_kernel<<<(256 * 8 * 32 * 16) / 256, 256, 0, stream>>>(Wf, Wb, Wp, wpack);
    pack_u_kernel<<<(96 * 16 * 32 * 16) / 256, 256, 0, stream>>>(Uf, upf);
    pack_u_kernel<<<(96 * 16 * 32 * 16) / 256, 256, 0, stream>>>(Ub, upb);
    proj_kernel  <<<dim3(4096, 8), 256, 0, stream>>>(xbf, wpack, bfv, bbv, proj);
    gru_kernel   <<<8, 512, 147456, stream>>>(proj, upf, upb, bfv, bbv, gruY);
    ln_kernel    <<<BBB * TT, 256, 0, stream>>>(gruY, proj, gamma, beta, out);
}
